// WaveFieldAttention_14920716386360
// MI455X (gfx1250) — compile-verified, matrix operand order checked by reading, not run on hardware
//
#include <hip/hip_runtime.h>
#include <hip/hip_bf16.h>
#include <math.h>

#define Bsz 4
#define Nn  2048
#define Dd  1024
#define Hh  16
#define dd  64
#define Ff  2048
#define PADN 8192
#define FBn 4097
#define NCn 32
#define EPSf 1e-6f
#define PI_F 3.14159265358979323846f

typedef __attribute__((ext_vector_type(16))) _Float16 v16h;
typedef __attribute__((ext_vector_type(8)))  float    v8f;

__device__ __forceinline__ float2 cmulf(float2 a, float2 b) {
  return make_float2(a.x*b.x - a.y*b.y, a.x*b.y + a.y*b.x);
}
__device__ __forceinline__ float sigmoidf_(float x) { return 1.0f / (1.0f + __expf(-x)); }
__device__ __forceinline__ float gelu_tanh(float x) {
  return 0.5f * x * (1.0f + tanhf(0.7978845608028654f * (x + 0.044715f * x * x * x)));
}

// ---------------------------------------------------------------------------
// WMMA GEMM:  C[m,n] = sum_k A[m,k] * Bw[n,k] + bias[n]
// A: MxK row-major f32, Bw: NxK row-major f32 (computes A @ Bw^T + bias).
// Block 256 threads (8 waves), 128x128 tile, BK=32.
// Wave tile 32x64 -> 8 x v_wmma_f32_16x16x32_f16 per K-step.
// LDS rows padded 32->36 halves to break bank conflicts on fragment reads.
// ---------------------------------------------------------------------------
#define LDP 36
__global__ void gemm_wmma_kernel(const float* __restrict__ A,
                                 const float* __restrict__ Bw,
                                 const float* __restrict__ bias,
                                 float* __restrict__ C,
                                 int M, int N, int K) {
  __shared__ _Float16 As[128][LDP];
  __shared__ _Float16 Bs[128][LDP];
  const int tid = threadIdx.x;
  const int m0 = blockIdx.y * 128;
  const int n0 = blockIdx.x * 128;
  const int wid = tid >> 5, lane = tid & 31;
  const int laneLo = lane & 15, laneHi = lane >> 4;
  const int wr = wid & 3;        // M strip of 32 rows
  const int wc = wid >> 2;       // N half of 64 cols
  v8f acc[2][4];
#pragma unroll
  for (int mi = 0; mi < 2; ++mi)
#pragma unroll
    for (int ni = 0; ni < 4; ++ni) acc[mi][ni] = (v8f){};

  for (int kb = 0; kb < K; kb += 32) {
    __syncthreads();
    // stage A/B tiles (f32 -> f16), 4 float4 per thread per tile
#pragma unroll
    for (int u = 0; u < 4; ++u) {
      int i = tid + u * 256;          // 0..1023
      int r = i >> 3, c4 = (i & 7) * 4;
      float4 va = *(const float4*)&A[(size_t)(m0 + r) * K + kb + c4];
      float4 vb = *(const float4*)&Bw[(size_t)(n0 + r) * K + kb + c4];
      union { _Float16 h[4]; uint2 u2; } pa, pb;
      pa.h[0] = (_Float16)va.x; pa.h[1] = (_Float16)va.y;
      pa.h[2] = (_Float16)va.z; pa.h[3] = (_Float16)va.w;
      pb.h[0] = (_Float16)vb.x; pb.h[1] = (_Float16)vb.y;
      pb.h[2] = (_Float16)vb.z; pb.h[3] = (_Float16)vb.w;
      *(uint2*)&As[r][c4] = pa.u2;
      *(uint2*)&Bs[r][c4] = pb.u2;
      if (kb + 32 < K) {              // prefetch next K tile (global_prefetch_b8)
        __builtin_prefetch(&A[(size_t)(m0 + r) * K + kb + 32 + c4], 0, 1);
        __builtin_prefetch(&Bw[(size_t)(n0 + r) * K + kb + 32 + c4], 0, 1);
      }
    }
    __syncthreads();

    union { v16h v; _Float16 e[16]; } ua[2], ub[4];
#pragma unroll
    for (int mi = 0; mi < 2; ++mi) {
      const _Float16* ar = As[wr * 32 + mi * 16 + laneLo];
#pragma unroll
      for (int el = 0; el < 16; ++el) {
        int vv = el >> 1, hf = el & 1;  // ISA 7.12.2 16-bit A 16x32 striping
        int kk = ((vv >> 2) << 4) + (laneHi << 3) + ((vv & 3) << 1) + hf;
        ua[mi].e[el] = ar[kk];
      }
    }
#pragma unroll
    for (int ni = 0; ni < 4; ++ni) {
      const _Float16* br = Bs[wc * 64 + ni * 16 + laneLo];
#pragma unroll
      for (int el = 0; el < 16; ++el) {
        int vv = el >> 1, hf = el & 1;
        int kk = ((vv >> 2) << 4) + (laneHi << 3) + ((vv & 3) << 1) + hf;
        ub[ni].e[el] = br[kk];
      }
    }
#pragma unroll
    for (int mi = 0; mi < 2; ++mi)
#pragma unroll
      for (int ni = 0; ni < 4; ++ni)
        acc[mi][ni] = __builtin_amdgcn_wmma_f32_16x16x32_f16(
            false, ua[mi].v, false, ub[ni].v, (short)0, acc[mi][ni], false, false);
  }

#pragma unroll
  for (int mi = 0; mi < 2; ++mi)
#pragma unroll
    for (int ni = 0; ni < 4; ++ni)
#pragma unroll
      for (int r = 0; r < 8; ++r) {
        int m = m0 + wr * 32 + mi * 16 + r + laneHi * 8;  // C/D: VGPR r -> M=r/r+8
        int n = n0 + wc * 64 + ni * 16 + laneLo;
        C[(size_t)m * N + n] = acc[mi][ni][r] + bias[n];
      }
}

// ---------------------------------------------------------------------------
// Feature maps + write-gate + field scatter.
// One block = 32 tokens of one (b,h). qkvg row layout: [q | k | v | g], 4096.
// ---------------------------------------------------------------------------
__global__ void featmap_field_kernel(const float* __restrict__ qkvg,
                                     const float* __restrict__ qfm_w, const float* __restrict__ qfm_b,
                                     const float* __restrict__ kfm_w, const float* __restrict__ kfm_b,
                                     const float* __restrict__ wg_w,  const float* __restrict__ wg_b,
                                     float* __restrict__ phi_q, float* __restrict__ field) {
  __shared__ float X[32][64];
  __shared__ float T[32][64];
  __shared__ float P[32][64];
  __shared__ float WG[32];
  const int bid = blockIdx.x;
  const int tblk = bid & 63;
  const int h = (bid >> 6) & 15;
  const int b = bid >> 10;
  const int n0 = tblk * 32;
  const int tid = threadIdx.x;
  const int tok = tid >> 3, l8 = tid & 7;

  const float strd = (float)(Ff - 1) / (float)(Nn - 1);

  // ---- phi_q ----
  for (int idx = tid; idx < 32 * 64; idx += 256) {
    int tk = idx >> 6, c = idx & 63;
    X[tk][c] = qkvg[(size_t)(b * Nn + n0 + tk) * (4 * Dd) + h * 64 + c];
  }
  __syncthreads();
#pragma unroll
  for (int jj = 0; jj < 8; ++jj) {
    int j = l8 * 8 + jj;
    float acc = qfm_b[j];
    for (int i = 0; i < 64; ++i) acc += X[tok][i] * qfm_w[j * 64 + i];
    T[tok][j] = fmaxf(acc, 0.0f);
  }
  __syncthreads();
#pragma unroll
  for (int jj = 0; jj < 8; ++jj) {
    int j = l8 * 8 + jj;
    float acc = qfm_b[64 + j];
    for (int i = 0; i < 64; ++i) acc += T[tok][i] * qfm_w[4096 + j * 64 + i];
    float p = fmaxf(acc, 0.0f) + EPSf;
    phi_q[((size_t)(b * Hh + h) * Nn + n0 + tok) * 64 + j] = p;
  }
  __syncthreads();

  // ---- phi_k + wg ----
  for (int idx = tid; idx < 32 * 64; idx += 256) {
    int tk = idx >> 6, c = idx & 63;
    X[tk][c] = qkvg[(size_t)(b * Nn + n0 + tk) * (4 * Dd) + Dd + h * 64 + c];
  }
  __syncthreads();
  if (tid < 32) {
    float acc = wg_b[0];
    for (int i = 0; i < 64; ++i) acc += X[tid][i] * wg_w[i];
    WG[tid] = sigmoidf_(acc);
  }
#pragma unroll
  for (int jj = 0; jj < 8; ++jj) {
    int j = l8 * 8 + jj;
    float acc = kfm_b[j];
    for (int i = 0; i < 64; ++i) acc += X[tok][i] * kfm_w[j * 64 + i];
    T[tok][j] = fmaxf(acc, 0.0f);
  }
  __syncthreads();
#pragma unroll
  for (int jj = 0; jj < 8; ++jj) {
    int j = l8 * 8 + jj;
    float acc = kfm_b[64 + j];
    for (int i = 0; i < 64; ++i) acc += T[tok][i] * kfm_w[4096 + j * 64 + i];
    P[tok][j] = fmaxf(acc, 0.0f) + EPSf;
  }
  __syncthreads();

  // ---- field scatter ----
  const int n = n0 + tok;
  float pos = fminf(fmaxf((float)n * strd, 0.0f), (float)(Ff - 2));
  int lo = min(max((int)pos, 0), Ff - 2);
  float frac = fminf(fmaxf(pos - (float)lo, 0.0f), 1.0f);
  int hi = lo + 1;
  float wgv = WG[tok];
  size_t fbase = (size_t)(b * Hh + h) * Ff * 128;
#pragma unroll
  for (int cc = 0; cc < 16; ++cc) {
    int c = l8 * 16 + cc;
    float w;
    if (c < 64) {
      float v = qkvg[(size_t)(b * Nn + n) * (4 * Dd) + 2 * Dd + h * 64 + c];
      w = wgv * P[tok][c] * v;
    } else {
      w = wgv * P[tok][c - 64];
    }
    atomicAdd(&field[fbase + (size_t)lo * 128 + c], w * (1.0f - frac));
    atomicAdd(&field[fbase + (size_t)hi * 128 + c], w * frac);
  }
}

// ---------------------------------------------------------------------------
// Control net: LN(q0) -> gelu MLP -> ctrl (B,H,NC). One block per batch.
// ---------------------------------------------------------------------------
__global__ void control_kernel(const float* __restrict__ qkvg,
                               const float* __restrict__ ln_g, const float* __restrict__ ln_b,
                               const float* __restrict__ sg_w1, const float* __restrict__ sg_b1,
                               const float* __restrict__ sg_w2, const float* __restrict__ sg_b2,
                               float* __restrict__ ctrl) {
  __shared__ float q0[1024];
  __shared__ float h1[1024];
  __shared__ float mu[16], iv[16];
  const int b = blockIdx.x;
  const int tid = threadIdx.x;
  for (int j = tid; j < 1024; j += 256)
    q0[j] = qkvg[(size_t)(b * Nn) * (4 * Dd) + j];     // q section, token 0
  __syncthreads();
  if (tid < 16) {
    float s = 0.f, s2 = 0.f;
    for (int i = 0; i < 64; ++i) { float v = q0[tid * 64 + i]; s += v; s2 += v * v; }
    float m = s / 64.0f;
    float var = s2 / 64.0f - m * m;
    mu[tid] = m;
    iv[tid] = rsqrtf(var + 1e-5f);
  }
  __syncthreads();
  for (int j = tid; j < 1024; j += 256) {
    int h = j >> 6, c = j & 63;
    q0[j] = (q0[j] - mu[h]) * iv[h] * ln_g[c] + ln_b[c];
  }
  __syncthreads();
  for (int j = tid; j < 1024; j += 256) {
    float acc = sg_b1[j];
    for (int i = 0; i < 1024; ++i) acc += q0[i] * sg_w1[(size_t)j * 1024 + i];
    h1[j] = gelu_tanh(acc);
  }
  __syncthreads();
  for (int j = tid; j < Hh * NCn; j += 256) {
    float acc = sg_b2[j];
    for (int i = 0; i < 1024; ++i) acc += h1[i] * sg_w2[(size_t)j * 1024 + i];
    ctrl[b * (Hh * NCn) + j] = acc;
  }
}

// ---------------------------------------------------------------------------
// In-LDS 8192-pt radix-2 FFT. Forward: DIF (natural -> bit-reversed).
// Inverse: DIT with conjugate twiddles (bit-reversed -> natural), unscaled.
// 256 threads.
// ---------------------------------------------------------------------------
__device__ void fft8192(float2* s, int tid, int inverse) {
  if (!inverse) {
    for (int st = 12; st >= 0; --st) {
      int m = 1 << st;
      __syncthreads();
      for (int j = tid; j < 4096; j += 256) {
        int k = j & (m - 1);
        int i = ((j >> st) << (st + 1)) + k;
        float2 a = s[i], bb = s[i + m];
        float sn, cs;
        __sincosf(-PI_F * (float)k / (float)m, &sn, &cs);
        float2 t = make_float2(a.x - bb.x, a.y - bb.y);
        s[i] = make_float2(a.x + bb.x, a.y + bb.y);
        s[i + m] = make_float2(t.x * cs - t.y * sn, t.x * sn + t.y * cs);
      }
    }
  } else {
    for (int st = 0; st <= 12; ++st) {
      int m = 1 << st;
      __syncthreads();
      for (int j = tid; j < 4096; j += 256) {
        int k = j & (m - 1);
        int i = ((j >> st) << (st + 1)) + k;
        float sn, cs;
        __sincosf(PI_F * (float)k / (float)m, &sn, &cs);
        float2 a = s[i], bb = s[i + m];
        float2 t = make_float2(bb.x * cs - bb.y * sn, bb.x * sn + bb.y * cs);
        s[i] = make_float2(a.x + t.x, a.y + t.y);
        s[i + m] = make_float2(a.x - t.x, a.y - t.y);
      }
    }
  }
  __syncthreads();
}

// base_fft[h] = FFT_8192(kern[h]), stored bit-reversed, full spectrum.
__global__ void basefft_kernel(const float* __restrict__ wave_freq,
                               const float* __restrict__ wave_damp,
                               const float* __restrict__ wave_phase,
                               float2* __restrict__ baseSpec) {
  extern __shared__ float2 z[];
  const int h = blockIdx.x;
  const int tid = threadIdx.x;
  const float dmp = __expf(wave_damp[h]);
  const float frq = wave_freq[h], ph = wave_phase[h];
  for (int i = tid; i < PADN; i += 256) {
    if (i < Ff) {
      float t = (float)i / (float)Ff;
      float kv = __expf(-dmp * t) * cosf(frq * t + ph);
      z[i] = make_float2(kv, 0.0f);
    } else {
      z[i] = make_float2(0.0f, 0.0f);
    }
  }
  fft8192(z, tid, 0);
  for (int i = tid; i < PADN; i += 256) baseSpec[(size_t)h * PADN + i] = z[i];
}

// modSpec[b,h,j] (bit-reversed order) = baseSpec[h,j]*(1+gate[b,h,fb])*(1/8192)
__global__ void mod_kernel(const float2* __restrict__ baseSpec,
                           const float* __restrict__ ctrl,
                           float2* __restrict__ modSpec) {
  int gid = blockIdx.x * 256 + threadIdx.x;           // over B*H*PADN
  int j = gid & (PADN - 1);
  int bh = gid >> 13;
  int h = bh & 15, b = bh >> 4;
  int f = (int)(__brev((unsigned)j) >> 19);           // bitrev13
  int fb = (f <= PADN / 2) ? f : (PADN - f);          // hermitian fold (0..4096)
  float ip = (float)fb * (float)(NCn - 1) / (float)(FBn - 1);
  int il = min(max((int)ip, 0), NCn - 2);
  float iw = ip - (float)il;
  const float* cb = ctrl + (b * Hh + h) * NCn;
  float gate = cb[il] * (1.0f - iw) + cb[il + 1] * iw;
  float sc = (1.0f + gate) * (1.0f / (float)PADN);
  float2 base = baseSpec[(size_t)h * PADN + j];
  modSpec[(size_t)(b * Hh + h) * PADN + j] = make_float2(base.x * sc, base.y * sc);
}

// Spectral convolution: two real channels packed per complex FFT.
__global__ void conv_kernel(const float* __restrict__ field,
                            const float2* __restrict__ modSpec,
                            float* __restrict__ conv) {
  extern __shared__ float2 z[];
  const int pair = blockIdx.x & 63;
  const int bh = blockIdx.x >> 6;
  const int c0 = pair * 2, c1 = c0 + 1;
  const int tid = threadIdx.x;
  for (int i = tid; i < PADN; i += 256) {
    if (i < Ff) {
      size_t base = ((size_t)bh * Ff + i) * 128;
      z[i] = make_float2(field[base + c0], field[base + c1]);
    } else {
      z[i] = make_float2(0.0f, 0.0f);
    }
  }
  fft8192(z, tid, 0);
  const float2* mp = modSpec + (size_t)bh * PADN;
  for (int j = tid; j < PADN; j += 256) z[j] = cmulf(z[j], mp[j]);
  fft8192(z, tid, 1);
  for (int i = tid; i < Ff; i += 256) {
    size_t base = ((size_t)bh * Ff + i) * 128;
    conv[base + c0] = z[i].x;
    conv[base + c1] = z[i].y;
  }
}

// Head coupling: convM[b,h,f,c] = sum_g coupling[h,g] * conv[b,g,f,c]
__global__ void couple_kernel(const float* __restrict__ conv,
                              const float* __restrict__ coupling,
                              float* __restrict__ convM) {
  __shared__ float cp[256];
  const int tid = threadIdx.x;
  cp[tid] = coupling[tid];
  __syncthreads();
  int gid = blockIdx.x * 256 + tid;                   // over B*F*128
  int c = gid & 127;
  int f = (gid >> 7) & (Ff - 1);
  int b = gid >> 18;
  float acc[16];
#pragma unroll
  for (int h = 0; h < 16; ++h) acc[h] = 0.0f;
  for (int g = 0; g < 16; ++g) {
    float v = conv[((size_t)(b * Hh + g) * Ff + f) * 128 + c];
#pragma unroll
    for (int h = 0; h < 16; ++h) acc[h] += cp[h * 16 + g] * v;
  }
#pragma unroll
  for (int h = 0; h < 16; ++h)
    convM[((size_t)(b * Hh + h) * Ff + f) * 128 + c] = acc[h];
}

// Gather + normalization gate -> attn (B,N,D). One 64-thread block per (b,n,h).
__global__ void gate_kernel(const float* __restrict__ convM,
                            const float* __restrict__ phi_q,
                            const float* __restrict__ qkvg,
                            float* __restrict__ attn) {
  __shared__ float red[64];
  const int bid = blockIdx.x;
  const int h = bid & 15;
  const int n = (bid >> 4) & (Nn - 1);
  const int b = bid >> 15;
  const int c = threadIdx.x;
  const float strd = (float)(Ff - 1) / (float)(Nn - 1);
  float pos = fminf(fmaxf((float)n * strd, 0.0f), (float)(Ff - 2));
  int lo = min(max((int)pos, 0), Ff - 2);
  float frac = fminf(fmaxf(pos - (float)lo, 0.0f), 1.0f);
  int hi = lo + 1;
  size_t base = (size_t)(b * Hh + h) * Ff;
  float ynum = convM[(base + lo) * 128 + c] * (1.0f - frac) +
               convM[(base + hi) * 128 + c] * frac;
  float yden = convM[(base + lo) * 128 + 64 + c] * (1.0f - frac) +
               convM[(base + hi) * 128 + 64 + c] * frac;
  float pq = phi_q[((size_t)(b * Hh + h) * Nn + n) * 64 + c];
  red[c] = pq * yden;
  __syncthreads();
  for (int s = 32; s > 0; s >>= 1) {
    if (c < s) red[c] += red[c + s];
    __syncthreads();
  }
  float den = fabsf(red[0]) + 1e-4f;
  float g = qkvg[(size_t)(b * Nn + n) * (4 * Dd) + 3 * Dd + h * 64 + c];
  attn[(size_t)(b * Nn + n) * Dd + h * 64 + c] = pq * ynum / den * sigmoidf_(g);
}

// ---------------------------------------------------------------------------
extern "C" void kernel_launch(void* const* d_in, const int* in_sizes, int n_in,
                              void* d_out, int out_size, void* d_ws, size_t ws_size,
                              hipStream_t stream) {
  (void)in_sizes; (void)n_in; (void)out_size; (void)ws_size;
  const float* x      = (const float*)d_in[0];
  const float* w_qkvg = (const float*)d_in[1];
  const float* b_qkvg = (const float*)d_in[2];
  const float* w_out  = (const float*)d_in[3];
  const float* b_out  = (const float*)d_in[4];
  const float* qfm_w  = (const float*)d_in[5];
  const float* qfm_b  = (const float*)d_in[6];
  const float* kfm_w  = (const float*)d_in[7];
  const float* kfm_b  = (const float*)d_in[8];
  const float* wg_w   = (const float*)d_in[9];
  const float* wg_b   = (const float*)d_in[10];
  const float* ln_g   = (const float*)d_in[11];
  const float* ln_b   = (const float*)d_in[12];
  const float* sg_w1  = (const float*)d_in[13];
  const float* sg_b1  = (const float*)d_in[14];
  const float* sg_w2  = (const float*)d_in[15];
  const float* sg_b2  = (const float*)d_in[16];
  const float* wfreq  = (const float*)d_in[17];
  const float* wdamp  = (const float*)d_in[18];
  const float* wphase = (const float*)d_in[19];
  const float* coup   = (const float*)d_in[20];
  float* out = (float*)d_out;

  char* ws = (char*)d_ws;
  size_t off = 0;
  auto carve = [&](size_t bytes) { char* p = ws + off; off += (bytes + 255) & ~(size_t)255; return p; };
  float*  qkvg    = (float*)carve((size_t)Bsz * Nn * 4 * Dd * 4);        // 128 MB
  float*  phi_q   = (float*)carve((size_t)Bsz * Hh * Nn * dd * 4);       // 32 MB
  float*  field   = (float*)carve((size_t)Bsz * Hh * Ff * 128 * 4);      // 64 MB
  float*  conv    = (float*)carve((size_t)Bsz * Hh * Ff * 128 * 4);      // 64 MB
  float*  convM   = (float*)carve((size_t)Bsz * Hh * Ff * 128 * 4);      // 64 MB
  float*  attn    = (float*)carve((size_t)Bsz * Nn * Dd * 4);            // 32 MB
  float2* baseSp  = (float2*)carve((size_t)Hh * PADN * 8);               // 1 MB
  float2* modSp   = (float2*)carve((size_t)Bsz * Hh * PADN * 8);         // 4 MB
  float*  ctrl    = (float*)carve((size_t)Bsz * Hh * NCn * 4);

  // 1) qkvg = x @ w_qkvg^T + b_qkvg   (8192x1024 @ 1024x4096)
  {
    dim3 grid(4 * Dd / 128, (Bsz * Nn) / 128);
    gemm_wmma_kernel<<<grid, 256, 0, stream>>>(x, w_qkvg, b_qkvg, qkvg,
                                               Bsz * Nn, 4 * Dd, Dd);
  }
  // 2) feature maps + field scatter
  hipMemsetAsync(field, 0, (size_t)Bsz * Hh * Ff * 128 * 4, stream);
  featmap_field_kernel<<<Bsz * Hh * (Nn / 32), 256, 0, stream>>>(
      qkvg, qfm_w, qfm_b, kfm_w, kfm_b, wg_w, wg_b, phi_q, field);
  // 3) control MLP
  control_kernel<<<Bsz, 256, 0, stream>>>(qkvg, ln_g, ln_b, sg_w1, sg_b1,
                                          sg_w2, sg_b2, ctrl);
  // 4) base kernel spectrum (bit-reversed)
  basefft_kernel<<<Hh, 256, PADN * sizeof(float2), stream>>>(wfreq, wdamp, wphase, baseSp);
  // 5) modulation spectrum
  mod_kernel<<<(Bsz * Hh * PADN) / 256, 256, 0, stream>>>(baseSp, ctrl, modSp);
  // 6) spectral convolution (FFT -> mul -> iFFT), 2 channels per FFT
  conv_kernel<<<Bsz * Hh * 64, 256, PADN * sizeof(float2), stream>>>(field, modSp, conv);
  // 7) head coupling
  couple_kernel<<<(Bsz * Ff * 128) / 256, 256, 0, stream>>>(conv, coup, convM);
  // 8) gather + gating -> attn
  gate_kernel<<<Bsz * Nn * Hh, 64, 0, stream>>>(convM, phi_q, qkvg, attn);
  // 9) out = attn @ w_out^T + b_out  (8192x1024 @ 1024x1024)
  {
    dim3 grid(Dd / 128, (Bsz * Nn) / 128);
    gemm_wmma_kernel<<<grid, 256, 0, stream>>>(attn, w_out, b_out, out,
                                               Bsz * Nn, Dd, Dd);
  }
}
